// CrossAttention_32469952757985
// MI455X (gfx1250) — matrix operand order repaired and checked by reading, not compiled, not run on hardware
//
#include <hip/hip_runtime.h>
#include <hip/hip_bf16.h>
#include <math.h>

#define D_MODEL 1024
#define NHEAD   8
#define HDIM    128
#define SEQ     2048
#define NROWS   8192   // B * SEQ

typedef __attribute__((ext_vector_type(16))) __bf16 v16bf;
typedef __attribute__((ext_vector_type(8)))  __bf16 v8bf;
typedef __attribute__((ext_vector_type(8)))  float  v8f;

union V16U { v16bf v; struct { v8bf lo, hi; } p; };

__device__ __forceinline__ v8f vzero8() {
  v8f z = {0.f,0.f,0.f,0.f,0.f,0.f,0.f,0.f};
  return z;
}

__device__ __forceinline__ v16bf frag_cat(const __bf16* p0, const __bf16* p1) {
  V16U u;
  u.p.lo = *(const v8bf*)p0;
  u.p.hi = *(const v8bf*)p1;
  return u.v;
}

__device__ __forceinline__ v8f wmma_bf16(v16bf a, v16bf b, v8f c) {
  return __builtin_amdgcn_wmma_f32_16x16x32_bf16(false, a, false, b, (short)0, c, false, false);
}

// A fragment 16x32 bf16: per-lane row base must already include lane row + 16*(lane>>4).
// frag = { base[k0 .. k0+7], base[k0+8 .. k0+15] }
__device__ __forceinline__ v16bf afrag_p(const __bf16* rowbase, int k0) {
  return frag_cat(rowbase + k0, rowbase + k0 + 8);
}
// B fragment 32x16 bf16 (transposed source): per-lane base includes col row + 16*(lane>>4).
// frag = { base[k0 .. k0+7], base[k0+8 .. k0+15] }
__device__ __forceinline__ v16bf bfrag_p(const __bf16* colbase, int k0) {
  return frag_cat(colbase + k0, colbase + k0 + 8);
}

__device__ __forceinline__ float redmax16(float v) {
  v = fmaxf(v, __shfl_xor(v, 1));
  v = fmaxf(v, __shfl_xor(v, 2));
  v = fmaxf(v, __shfl_xor(v, 4));
  v = fmaxf(v, __shfl_xor(v, 8));
  return v;
}
__device__ __forceinline__ float redsum16(float v) {
  v += __shfl_xor(v, 1);
  v += __shfl_xor(v, 2);
  v += __shfl_xor(v, 4);
  v += __shfl_xor(v, 8);
  return v;
}

// ---------------- weight transpose: Wt[n][k] = bf16(W[k][n]) ----------------
__global__ void wtrans_k(const float* __restrict__ W, __bf16* __restrict__ Wt) {
  int idx = blockIdx.x * 256 + threadIdx.x;      // over 1024*1024, coalesced read
  int kk = idx >> 10;
  int n  = idx & 1023;
  Wt[(long)n * D_MODEL + kk] = (__bf16)W[idx];
}

// ---------------- row layernorm (1024 cols) -> bf16 (or plain convert) ------
__global__ void rows_ln_k(const float* __restrict__ X, const float* __restrict__ g,
                          const float* __restrict__ b, __bf16* __restrict__ Y, int doLN) {
  __shared__ float s1[256];
  __shared__ float s2[256];
  int row = blockIdx.x, t = threadIdx.x;
  const float* x = X + (long)row * D_MODEL;
  __bf16* y = Y + (long)row * D_MODEL;
  float v0 = x[t], v1 = x[t + 256], v2 = x[t + 512], v3 = x[t + 768];
  if (!doLN) {
    y[t] = (__bf16)v0; y[t + 256] = (__bf16)v1;
    y[t + 512] = (__bf16)v2; y[t + 768] = (__bf16)v3;
    return;
  }
  s1[t] = v0 + v1 + v2 + v3;
  s2[t] = v0 * v0 + v1 * v1 + v2 * v2 + v3 * v3;
  __syncthreads();
  for (int off = 128; off > 0; off >>= 1) {
    if (t < off) { s1[t] += s1[t + off]; s2[t] += s2[t + off]; }
    __syncthreads();
  }
  float mu  = s1[0] * (1.0f / 1024.0f);
  float var = s2[0] * (1.0f / 1024.0f) - mu * mu;
  float rs  = rsqrtf(var + 1e-5f);
  y[t]       = (__bf16)((v0 - mu) * rs * g[t]       + b[t]);
  y[t + 256] = (__bf16)((v1 - mu) * rs * g[t + 256] + b[t + 256]);
  y[t + 512] = (__bf16)((v2 - mu) * rs * g[t + 512] + b[t + 512]);
  y[t + 768] = (__bf16)((v3 - mu) * rs * g[t + 768] + b[t + 768]);
}

// ---------------- projection GEMM: Y = X(8192x1024) * Wt^T, bf16 WMMA -------
// Wave tile 32x64 (2 A frags x 4 B frags). Per-lane base pointers hoisted; K loop
// unrolled x4 so fragments become base+immediate and the scheduler pipelines.
// vt_mode 0: store [B,H,L,HD] (q,k) ; vt_mode 1: store [B,H,HD,L] (v^T)
__global__ void gemm_proj_k(const __bf16* __restrict__ X, const __bf16* __restrict__ Wt,
                            __bf16* __restrict__ Y, int vt_mode) {
  int lane = threadIdx.x & 31;
  int wave = threadIdx.x >> 5;
  int half = (lane >> 4) & 1;
  int rbase = blockIdx.x * 128 + (wave & 3) * 32;
  int nbase = blockIdx.y * 128 + (wave >> 2) * 64;

  const __bf16* a0p = X + (long)(rbase + (lane & 15)) * D_MODEL + 16 * half;
  const __bf16* a1p = a0p + 16 * D_MODEL;
  const __bf16* bbase = Wt + (long)(nbase + (lane & 15)) * D_MODEL + 16 * half;
  const __bf16* b0p = bbase;
  const __bf16* b1p = bbase + 16 * D_MODEL;
  const __bf16* b2p = bbase + 32 * D_MODEL;
  const __bf16* b3p = bbase + 48 * D_MODEL;

  v8f acc[2][4];
#pragma unroll
  for (int i = 0; i < 2; ++i)
#pragma unroll
    for (int j = 0; j < 4; ++j) acc[i][j] = vzero8();

#pragma unroll 4
  for (int k0 = 0; k0 < D_MODEL; k0 += 32) {
    v16bf a0 = afrag_p(a0p, k0);
    v16bf a1 = afrag_p(a1p, k0);
    v16bf b0 = bfrag_p(b0p, k0);
    v16bf b1 = bfrag_p(b1p, k0);
    v16bf b2 = bfrag_p(b2p, k0);
    v16bf b3 = bfrag_p(b3p, k0);
    acc[0][0] = wmma_bf16(a0, b0, acc[0][0]);
    acc[1][0] = wmma_bf16(a1, b0, acc[1][0]);
    acc[0][1] = wmma_bf16(a0, b1, acc[0][1]);
    acc[1][1] = wmma_bf16(a1, b1, acc[1][1]);
    acc[0][2] = wmma_bf16(a0, b2, acc[0][2]);
    acc[1][2] = wmma_bf16(a1, b2, acc[1][2]);
    acc[0][3] = wmma_bf16(a0, b3, acc[0][3]);
    acc[1][3] = wmma_bf16(a1, b3, acc[1][3]);
  }

  int mlo  = 8 * (lane >> 4);
  int ncol = lane & 15;
#pragma unroll
  for (int i = 0; i < 2; ++i) {
#pragma unroll
    for (int j = 0; j < 4; ++j) {
      int col = nbase + 16 * j + ncol;
      int hh = col >> 7, d = col & (HDIM - 1);
#pragma unroll
      for (int r = 0; r < 8; ++r) {
        int row = rbase + 16 * i + mlo + r;
        int bb = row >> 11, ll = row & (SEQ - 1);
        long idx;
        if (vt_mode) idx = ((long)((bb * NHEAD + hh) * HDIM + d)) * SEQ + ll;
        else         idx = ((long)((bb * NHEAD + hh) * SEQ + ll)) * HDIM + d;
        Y[idx] = (__bf16)acc[i][j][r];
      }
    }
  }
}

// ---------------- flash attention: one wave per 16-row q tile ---------------
__global__ void attn_k(const __bf16* __restrict__ q, const __bf16* __restrict__ k,
                       const __bf16* __restrict__ vT, float* __restrict__ O) {
  __shared__ __align__(16) __bf16 Pt[8][16 * 32];   // per-wave P staging
  int lane = threadIdx.x & 31;
  int wave = threadIdx.x >> 5;
  int half = (lane >> 4) & 1;
  int bh = blockIdx.x;
  int bb = bh >> 3, hh = bh & (NHEAD - 1);
  int qrow = blockIdx.y * 128 + wave * 16;

  const __bf16* qp = q  + (long)bh * SEQ * HDIM;   // [L][HD]
  const __bf16* kp = k  + (long)bh * SEQ * HDIM;   // [L][HD]
  const __bf16* vp = vT + (long)bh * HDIM * SEQ;   // [HD][L]
  __bf16* P = Pt[wave];

  // per-lane base pointers (loop-invariant)
  const __bf16* qrp = qp + (long)(qrow + (lane & 15)) * HDIM + 16 * half;  // A rows of q
  const __bf16* krp = kp + (long)(lane & 15) * HDIM + 16 * half;          // B cols (l) of k^T
  const __bf16* vrp = vp + (long)(lane & 15) * SEQ + 16 * half;           // B cols (d) of v

  v16bf qa[4];
#pragma unroll
  for (int c = 0; c < 4; ++c) qa[c] = afrag_p(qrp, 32 * c);

  v8f acc[8];
#pragma unroll
  for (int j = 0; j < 8; ++j) acc[j] = vzero8();
  float rmaxv[8], rsumv[8], alpha[8];
#pragma unroll
  for (int r = 0; r < 8; ++r) { rmaxv[r] = -3.0e38f; rsumv[r] = 0.0f; }

  const float INVT = 0.03125f;  // 1/sqrt(1024)

#pragma unroll 1
  for (int kt = 0; kt < SEQ; kt += 32) {
    const __bf16* kt0 = krp + (long)kt * HDIM;        // l = kt + (lane&15)
    const __bf16* kt1 = kt0 + 16 * HDIM;              // l = kt+16 + (lane&15)

    // ---- scores: S = q x k^T (body is straight-line; scheduler hoists loads)
    v8f s0 = vzero8(), s1 = vzero8();
#pragma unroll
    for (int c = 0; c < 4; ++c) {
      v16bf bk0 = bfrag_p(kt0, 32 * c);
      v16bf bk1 = bfrag_p(kt1, 32 * c);
      s0 = wmma_bf16(qa[c], bk0, s0);
      s1 = wmma_bf16(qa[c], bk1, s1);
    }

    // ---- online softmax ----
#pragma unroll
    for (int r = 0; r < 8; ++r) {
      float a0 = fminf(fmaxf(s0[r] * INVT, -10000.f), 10000.f);
      float a1 = fminf(fmaxf(s1[r] * INVT, -10000.f), 10000.f);
      float mx   = redmax16(fmaxf(a0, a1));
      float mnew = fmaxf(rmaxv[r], mx);
      alpha[r]   = __expf(rmaxv[r] - mnew);
      rmaxv[r]   = mnew;
      float p0 = __expf(a0 - mnew);
      float p1 = __expf(a1 - mnew);
      rsumv[r] = rsumv[r] * alpha[r] + redsum16(p0 + p1);
      int m = 8 * (lane >> 4) + r;
      P[m * 32 +      (lane & 15)] = (__bf16)p0;
      P[m * 32 + 16 + (lane & 15)] = (__bf16)p1;
    }
#pragma unroll
    for (int j = 0; j < 8; ++j)
#pragma unroll
      for (int r = 0; r < 8; ++r) acc[j][r] *= alpha[r];

    // preload first v-fragments so they fly during the LDS turnaround
    v16bf vb0 = frag_cat(vrp + kt, vrp + kt + 8);
    v16bf vb1 = frag_cat(vrp + 16 * SEQ + kt, vrp + 16 * SEQ + kt + 8);

    // wave-private LDS RAW: HW LDS pipe is in-order per wave; fence compiler + wait
    asm volatile("s_wait_dscnt 0x0" ::: "memory");
    const __bf16* pr = P + (lane & 15) * 32 + 16 * half;
    v16bf pa = frag_cat(pr, pr + 8);
    asm volatile("" ::: "memory");   // keep next-iter LDS writes after these reads

    // ---- PV: O += P x v ----
    acc[0] = wmma_bf16(pa, vb0, acc[0]);
    acc[1] = wmma_bf16(pa, vb1, acc[1]);
#pragma unroll
    for (int j = 2; j < 8; ++j) {
      const __bf16* vj = vrp + (long)(16 * j) * SEQ + kt;
      v16bf bv = frag_cat(vj, vj + 8);
      acc[j] = wmma_bf16(pa, bv, acc[j]);
    }
  }

#pragma unroll
  for (int r = 0; r < 8; ++r) alpha[r] = 1.0f / rsumv[r];
  int mlo = 8 * (lane >> 4);
#pragma unroll
  for (int j = 0; j < 8; ++j) {
    int d = 16 * j + (lane & 15);
#pragma unroll
    for (int r = 0; r < 8; ++r) {
      int row = qrow + mlo + r;
      O[((long)(bb * SEQ + row)) * D_MODEL + hh * HDIM + d] = acc[j][r] * alpha[r];
    }
  }
}

// ---------------- final GEMM: out = On + gelu(On @ Wo), f32 out -------------
__global__ void gemm_final_k(const __bf16* __restrict__ On, const __bf16* __restrict__ Wt,
                             float* __restrict__ out) {
  int lane = threadIdx.x & 31;
  int wave = threadIdx.x >> 5;
  int half = (lane >> 4) & 1;
  int rbase = blockIdx.x * 128 + (wave & 3) * 32;
  int nbase = blockIdx.y * 128 + (wave >> 2) * 64;

  const __bf16* a0p = On + (long)(rbase + (lane & 15)) * D_MODEL + 16 * half;
  const __bf16* a1p = a0p + 16 * D_MODEL;
  const __bf16* bbase = Wt + (long)(nbase + (lane & 15)) * D_MODEL + 16 * half;
  const __bf16* b0p = bbase;
  const __bf16* b1p = bbase + 16 * D_MODEL;
  const __bf16* b2p = bbase + 32 * D_MODEL;
  const __bf16* b3p = bbase + 48 * D_MODEL;

  v8f acc[2][4];
#pragma unroll
  for (int i = 0; i < 2; ++i)
#pragma unroll
    for (int j = 0; j < 4; ++j) acc[i][j] = vzero8();

#pragma unroll 4
  for (int k0 = 0; k0 < D_MODEL; k0 += 32) {
    v16bf a0 = afrag_p(a0p, k0);
    v16bf a1 = afrag_p(a1p, k0);
    v16bf b0 = bfrag_p(b0p, k0);
    v16bf b1 = bfrag_p(b1p, k0);
    v16bf b2 = bfrag_p(b2p, k0);
    v16bf b3 = bfrag_p(b3p, k0);
    acc[0][0] = wmma_bf16(a0, b0, acc[0][0]);
    acc[1][0] = wmma_bf16(a1, b0, acc[1][0]);
    acc[0][1] = wmma_bf16(a0, b1, acc[0][1]);
    acc[1][1] = wmma_bf16(a1, b1, acc[1][1]);
    acc[0][2] = wmma_bf16(a0, b2, acc[0][2]);
    acc[1][2] = wmma_bf16(a1, b2, acc[1][2]);
    acc[0][3] = wmma_bf16(a0, b3, acc[0][3]);
    acc[1][3] = wmma_bf16(a1, b3, acc[1][3]);
  }

  int mlo  = 8 * (lane >> 4);
  int ncol = lane & 15;
#pragma unroll
  for (int i = 0; i < 2; ++i) {
#pragma unroll
    for (int j = 0; j < 4; ++j) {
      int col = nbase + 16 * j + ncol;
#pragma unroll
      for (int r = 0; r < 8; ++r) {
        int row = rbase + 16 * i + mlo + r;
        float y   = acc[i][j][r];
        float res = (float)On[(long)row * D_MODEL + col];
        float ge  = 0.5f * y * (1.0f + erff(y * 0.70710678118f));
        out[(long)row * D_MODEL + col] = res + ge;
      }
    }
  }
}

extern "C" void kernel_launch(void* const* d_in, const int* in_sizes, int n_in,
                              void* d_out, int out_size, void* d_ws, size_t ws_size,
                              hipStream_t stream) {
  (void)in_sizes; (void)n_in; (void)out_size; (void)ws_size;
  const float* Q     = (const float*)d_in[0];
  const float* Kc    = (const float*)d_in[1];
  const float* V     = (const float*)d_in[2];
  const float* Wq    = (const float*)d_in[3];
  const float* Wk    = (const float*)d_in[4];
  const float* Wv    = (const float*)d_in[5];
  const float* Wo    = (const float*)d_in[6];
  const float* pre_g = (const float*)d_in[7];
  const float* pre_b = (const float*)d_in[8];
  const float* ln_g  = (const float*)d_in[9];
  const float* ln_b  = (const float*)d_in[10];
  float* out = (float*)d_out;

  char* ws = (char*)d_ws;
  const size_t MB = 1ull << 20;
  __bf16* WqT = (__bf16*)(ws + 0 * MB);    // 2 MB each
  __bf16* WkT = (__bf16*)(ws + 2 * MB);
  __bf16* WvT = (__bf16*)(ws + 4 * MB);
  __bf16* WoT = (__bf16*)(ws + 6 * MB);
  __bf16* Qn  = (__bf16*)(ws + 8 * MB);    // 16 MB each
  __bf16* Kn  = (__bf16*)(ws + 24 * MB);
  __bf16* Vb  = (__bf16*)(ws + 40 * MB);
  __bf16* qh  = (__bf16*)(ws + 56 * MB);
  __bf16* kh  = (__bf16*)(ws + 72 * MB);
  __bf16* vTh = (__bf16*)(ws + 88 * MB);
  float*  Obuf = (float*)(ws + 8 * MB);    // 32 MB, aliases Qn/Kn (dead)
  __bf16* On   = (__bf16*)(ws + 56 * MB);  // aliases qh (dead after attention)

  wtrans_k<<<4096, 256, 0, stream>>>(Wq, WqT);
  wtrans_k<<<4096, 256, 0, stream>>>(Wk, WkT);
  wtrans_k<<<4096, 256, 0, stream>>>(Wv, WvT);
  wtrans_k<<<4096, 256, 0, stream>>>(Wo, WoT);

  rows_ln_k<<<NROWS, 256, 0, stream>>>(Q,  pre_g, pre_b, Qn, 1);
  rows_ln_k<<<NROWS, 256, 0, stream>>>(Kc, pre_g, pre_b, Kn, 1);
  rows_ln_k<<<NROWS, 256, 0, stream>>>(V,  pre_g, pre_b, Vb, 0);

  dim3 gg(NROWS / 128, D_MODEL / 128);
  gemm_proj_k<<<gg, 256, 0, stream>>>(Qn, WqT, qh, 0);
  gemm_proj_k<<<gg, 256, 0, stream>>>(Kn, WkT, kh, 0);
  gemm_proj_k<<<gg, 256, 0, stream>>>(Vb, WvT, vTh, 1);

  attn_k<<<dim3(4 * NHEAD, SEQ / 128), 256, 0, stream>>>(qh, kh, vTh, Obuf);

  rows_ln_k<<<NROWS, 256, 0, stream>>>(Obuf, ln_g, ln_b, On, 1);

  gemm_final_k<<<gg, 256, 0, stream>>>(On, WoT, out);
}
